// TtMambaBlock_24154896073599
// MI455X (gfx1250) — matrix operand order repaired and checked by reading, not compiled, not run on hardware
//
#include <hip/hip_runtime.h>

// ---- WMMA vector types (gfx1250, wave32) ----
typedef __attribute__((ext_vector_type(16))) __bf16 v16bf;
typedef __attribute__((ext_vector_type(8)))  __bf16 v8bf;
typedef __attribute__((ext_vector_type(8)))  float  v8f;
typedef __attribute__((ext_vector_type(4)))  float  v4f;
typedef __attribute__((ext_vector_type(4)))  int    v4i;

#define U_DIM   1024
#define H_DIM   4096
#define D_INNER 8192

#define BM 128
#define BN 128
#define BK 64
#define LDS_STRIDE 72            // BK + 8 bf16 pad: row stride 144B (16B aligned, bank-spread)
#define ATILE (BM * LDS_STRIDE)  // ushort elements per A buffer
#define BTILE (BN * LDS_STRIDE)
#define NK (H_DIM / BK)          // 64 K-steps

// ---- CDNA5 async global->LDS DMA probes ----
#if defined(__has_builtin)
#if __has_builtin(__builtin_amdgcn_global_load_async_to_lds_b128)
#define HAVE_ASYNC_LDS 1
#endif
#if __has_builtin(__builtin_amdgcn_s_wait_asynccnt)
#define HAVE_WAIT_ASYNC 1
#endif
#endif

static __device__ __forceinline__ void wait_async0() {
#if defined(HAVE_WAIT_ASYNC)
    __builtin_amdgcn_s_wait_asynccnt(0);
#else
    asm volatile("s_wait_asynccnt 0x0" ::: "memory");
#endif
}

// round-to-nearest-even f32 -> bf16 (bit trick; no cvt_pk builtin on this toolchain)
static __device__ __forceinline__ unsigned int f2bf(float f) {
    union { float f; unsigned int u; } v; v.f = f;
    unsigned int r = v.u + 0x7FFFu + ((v.u >> 16) & 1u);
    return r >> 16;
}
static __device__ __forceinline__ unsigned long long pack4bf(v4f a) {
    return (unsigned long long)(f2bf(a.x) | (f2bf(a.y) << 16))
         | ((unsigned long long)(f2bf(a.z) | (f2bf(a.w) << 16)) << 32);
}

// ---- pre-pass 1: f32 -> bf16 straight copy (for x), 4 elements/thread ----
__global__ __launch_bounds__(256)
void cvt_bf16_kernel(const float* __restrict__ src, unsigned short* __restrict__ dst, int n4) {
    const int i = blockIdx.x * blockDim.x + threadIdx.x;
    if (i < n4) {
        const v4f a = ((const v4f*)src)[i];
        ((unsigned long long*)dst)[i] = pack4bf(a);
    }
}

// ---- pre-pass 2: f32 [K][N] -> bf16 [N][K] (transpose for proj), 64x64 LDS tiles ----
__global__ __launch_bounds__(256)
void cvt_transpose_bf16_kernel(const float* __restrict__ src, unsigned short* __restrict__ dst,
                               int K, int N) {
    __shared__ unsigned short T[64 * LDS_STRIDE];
    const int tid = threadIdx.x;
    const int n0 = blockIdx.x * 64;
    const int k0 = blockIdx.y * 64;
    const int c4 = (tid & 15) * 4;
    const int r0 = tid >> 4;                  // 0..15
#pragma unroll
    for (int p = 0; p < 4; ++p) {             // read [k][n] coalesced, convert, stage
        const int r = r0 + p * 16;            // k-local
        const v4f a = *(const v4f*)&src[(size_t)(k0 + r) * N + n0 + c4];
        *(unsigned long long*)&T[r * LDS_STRIDE + c4] = pack4bf(a);
    }
    __syncthreads();
#pragma unroll
    for (int p = 0; p < 4; ++p) {             // write [n][k] coalesced
        const int nl = r0 + p * 16;           // n-local
        const unsigned long long v =
              (unsigned long long)T[(c4 + 0) * LDS_STRIDE + nl]
            | ((unsigned long long)T[(c4 + 1) * LDS_STRIDE + nl] << 16)
            | ((unsigned long long)T[(c4 + 2) * LDS_STRIDE + nl] << 32)
            | ((unsigned long long)T[(c4 + 3) * LDS_STRIDE + nl] << 48);
        *(unsigned long long*)&dst[(size_t)(n0 + nl) * K + k0 + c4] = v;
    }
}

// ---- main kernel: GEMM (bf16 WMMA, f32 accum) + fused conv-state epilogue ----
// PRE=true : Asrc = xbf [M][K] bf16, Bsrc = pbfT [N][K] bf16 -> async DMA staging (no VGPRs)
// PRE=false: Asrc/Bsrc are the original f32 tensors -> convert+transpose during staging
template <bool PRE>
__global__ __launch_bounds__(256, 1)
void mamba_inproj_conv_kernel(const void* __restrict__ Asrc,
                              const void* __restrict__ Bsrc,
                              const float* __restrict__ conv_states, // [4,1024,8192]
                              const float* __restrict__ conv_wts,    // [4,8192]
                              const float* __restrict__ conv_bias,   // [8192]
                              float* __restrict__ out)               // [out | new_states]
{
    extern __shared__ __align__(16) unsigned short smem[];  // [As0][As1][Bs0][Bs1] = 73728 B

    const int tid  = threadIdx.x;
    const int wave = tid >> 5;
    const int lane = tid & 31;
    const int l16  = lane & 15;
    const int hi   = lane >> 4;

    const int bn = blockIdx.x * BN;
    const int bm = blockIdx.y * BM;
    const int wm = (wave >> 2) * 64;
    const int wn = (wave & 3) * 32;

    v8f acc[4][2];
#pragma unroll
    for (int i = 0; i < 4; ++i)
#pragma unroll
        for (int j = 0; j < 2; ++j) acc[i][j] = (v8f)0.0f;

    // staging decompositions (one base address per array; all else immediate offsets)
    const int a_c4 = (tid & 15) * 4;   // fused path: 16 col-groups of 4 f32
    const int a_r0 = tid >> 4;         // fused path: rows, 8 passes of 16
    const int f_c8 = (tid & 7) * 8;    // fast path: 8 col-groups of 8 bf16
    const int f_r0 = tid >> 3;         // fast path: rows, 4 passes of 32
    const int b_n  = tid & 127;        // fused B: column (invariant across s: 256 % 128 == 0)
    const int b_kb = (tid >> 7) * 4;   // fused B: k base; k = b_kb + 8*s

    // stage tile (global -> LDS buffer `buf`) for K offset k0
    auto stage_tile = [&](int buf, int k0) {
        unsigned short* Aw = &smem[buf * ATILE];
        unsigned short* Bw = &smem[2 * ATILE + buf * BTILE];
        if constexpr (PRE) {
            const unsigned short* ap =
                (const unsigned short*)Asrc + (size_t)(bm + f_r0) * H_DIM + k0 + f_c8;
            const unsigned short* bp =
                (const unsigned short*)Bsrc + (size_t)(bn + f_r0) * H_DIM + k0 + f_c8;
            unsigned short* aw = &Aw[f_r0 * LDS_STRIDE + f_c8];
            unsigned short* bw = &Bw[f_r0 * LDS_STRIDE + f_c8];
#if defined(HAVE_ASYNC_LDS)
            // CDNA5 async DMA: global -> LDS, tracked by ASYNCcnt, zero staging VGPRs
#pragma unroll
            for (int p = 0; p < 4; ++p)
                __builtin_amdgcn_global_load_async_to_lds_b128(
                    (__attribute__((address_space(1))) v4i*)(ap + p * 32 * H_DIM),
                    (__attribute__((address_space(3))) v4i*)(aw + p * 32 * LDS_STRIDE),
                    0, 0);
#pragma unroll
            for (int p = 0; p < 4; ++p)
                __builtin_amdgcn_global_load_async_to_lds_b128(
                    (__attribute__((address_space(1))) v4i*)(bp + p * 32 * H_DIM),
                    (__attribute__((address_space(3))) v4i*)(bw + p * 32 * LDS_STRIDE),
                    0, 0);
#else
            // fallback: short-lived register relay (load -> immediate LDS store)
#pragma unroll
            for (int p = 0; p < 4; ++p) {
                const uint4 a = *(const uint4*)(ap + p * 32 * H_DIM);
                *(uint4*)(aw + p * 32 * LDS_STRIDE) = a;
            }
#pragma unroll
            for (int p = 0; p < 4; ++p) {
                const uint4 b = *(const uint4*)(bp + p * 32 * H_DIM);
                *(uint4*)(bw + p * 32 * LDS_STRIDE) = b;
            }
#endif
        } else {
            const float* xp_ =
                (const float*)Asrc + (size_t)(bm + a_r0) * H_DIM + k0 + a_c4;
            unsigned short* aw = &Aw[a_r0 * LDS_STRIDE + a_c4];
#pragma unroll
            for (int p = 0; p < 8; ++p) {
                const v4f a = *(const v4f*)(xp_ + p * 16 * H_DIM);
                *(unsigned long long*)(aw + p * 16 * LDS_STRIDE) = pack4bf(a);
            }
            const float* bp =
                (const float*)Bsrc + (size_t)(k0 + b_kb) * D_INNER + bn + b_n;
            unsigned short* bw = &Bw[b_n * LDS_STRIDE + b_kb];
#pragma unroll
            for (int s = 0; s < 8; ++s) {
                const float* q = bp + s * 8 * D_INNER;
                const v4f b = (v4f){q[0], q[D_INNER], q[2 * D_INNER], q[3 * D_INNER]};
                *(unsigned long long*)(bw + s * 8) = pack4bf(b);
            }
        }
    };

    // ---------- prologue ----------
    stage_tile(0, 0);
#if defined(HAVE_ASYNC_LDS)
    if constexpr (PRE) wait_async0();
#endif
    __syncthreads();

    // ---------- main pipeline: one barrier per K-step, double-buffered LDS ----------
#pragma unroll 1
    for (int it = 0; it < NK; ++it) {
        const int cur = it & 1;

        // fire next tile's copies into the other buffer (async: fire-and-forget)
        if (it + 1 < NK) stage_tile(cur ^ 1, (it + 1) * BK);

        const unsigned short* Ab = &smem[cur * ATILE];
        const unsigned short* Bb = &smem[2 * ATILE + cur * BTILE];
#pragma unroll
        for (int kk = 0; kk < 2; ++kk) {
            const int kb = kk * 32;
            v16bf afrag[4];
#pragma unroll
            for (int i = 0; i < 4; ++i) {
                const unsigned short* pa = &Ab[(wm + i * 16 + l16) * LDS_STRIDE + kb];
                // A striping: lanes 0-15 hold K{0..7,16..23}, lanes 16-31 K{8..15,24..31}
                v16bf f;
                ((v8bf*)&f)[0] = *(const v8bf*)&pa[hi * 8];
                ((v8bf*)&f)[1] = *(const v8bf*)&pa[16 + hi * 8];
                afrag[i] = f;
            }
            v16bf bfrag[2];
#pragma unroll
            for (int j = 0; j < 2; ++j) {
                const unsigned short* pb2 = &Bb[(wn + j * 16 + l16) * LDS_STRIDE + kb];
                // B (K-major per column): lanes 0-15 hold K{0..7,16..23}, lanes 16-31 K{8..15,24..31}
                v16bf f;
                ((v8bf*)&f)[0] = *(const v8bf*)&pb2[hi * 8];
                ((v8bf*)&f)[1] = *(const v8bf*)&pb2[16 + hi * 8];
                bfrag[j] = f;
            }
#pragma unroll
            for (int i = 0; i < 4; ++i)
#pragma unroll
                for (int j = 0; j < 2; ++j)
                    acc[i][j] = __builtin_amdgcn_wmma_f32_16x16x32_bf16(
                        false, afrag[i], false, bfrag[j],
                        (short)0, acc[i][j], false, false);
        }

#if defined(HAVE_ASYNC_LDS)
        if constexpr (PRE) wait_async0();   // my async copies for buf^1 are done
#endif
        __syncthreads();                    // everyone's copies visible
    }

    // ---------- fused epilogue: state shift + 4-tap depthwise conv + bias ----------
    const size_t SZ = (size_t)U_DIM * D_INNER;
#pragma unroll
    for (int j = 0; j < 2; ++j) {
        const int col = bn + wn + j * 16 + l16;
        const float w0 = conv_wts[0 * D_INNER + col];
        const float w1 = conv_wts[1 * D_INNER + col];
        const float w2 = conv_wts[2 * D_INNER + col];
        const float w3 = conv_wts[3 * D_INNER + col];
        const float bb = conv_bias[col];
#pragma unroll
        for (int i = 0; i < 4; ++i) {
#pragma unroll
            for (int r = 0; r < 8; ++r) {
                const int row = bm + wm + i * 16 + hi * 8 + r;  // C layout: M = r + 8*hi
                const size_t idx = (size_t)row * D_INNER + col;
                const float xp = acc[i][j][r];
                const float s1 = conv_states[1 * SZ + idx];
                const float s2 = conv_states[2 * SZ + idx];
                const float s3 = conv_states[3 * SZ + idx];
                out[SZ + 0 * SZ + idx] = s1;   // new_states[0] = conv_states[1]
                out[SZ + 1 * SZ + idx] = s2;   // new_states[1] = conv_states[2]
                out[SZ + 2 * SZ + idx] = s3;   // new_states[2] = conv_states[3]
                out[SZ + 3 * SZ + idx] = xp;   // new_states[3] = xp
                out[idx] = w0 * s1 + w1 * s2 + w2 * s3 + w3 * xp + bb;
            }
        }
    }
}

extern "C" void kernel_launch(void* const* d_in, const int* in_sizes, int n_in,
                              void* d_out, int out_size, void* d_ws, size_t ws_size,
                              hipStream_t stream) {
    (void)in_sizes; (void)n_in; (void)out_size;
    const float* x           = (const float*)d_in[0];
    const float* ssm_proj    = (const float*)d_in[1];
    const float* conv_states = (const float*)d_in[2];
    const float* conv_wts    = (const float*)d_in[3];
    const float* conv_bias   = (const float*)d_in[4];
    float* out = (float*)d_out;

    dim3 grid(D_INNER / BN, U_DIM / BM);   // (64, 8)
    dim3 block(256);                        // 8 wave32s
    const size_t smem_bytes = (size_t)(2 * ATILE + 2 * BTILE) * sizeof(unsigned short); // 73728

    const size_t xElems = (size_t)U_DIM * H_DIM;       // 4.19M
    const size_t pElems = (size_t)H_DIM * D_INNER;     // 33.6M
    const size_t need   = (xElems + pElems) * sizeof(unsigned short);  // ~75.5 MB

    if (ws_size >= need) {
        // fast path: one-time f32->bf16 (+transpose for proj) pre-pass into scratch
        unsigned short* xbf  = (unsigned short*)d_ws;      // [1024][4096] bf16
        unsigned short* pbfT = xbf + xElems;               // [8192][4096] bf16 (transposed)
        const int n4x = (int)(xElems / 4);
        cvt_bf16_kernel<<<dim3((n4x + 255) / 256), block, 0, stream>>>(x, xbf, n4x);
        cvt_transpose_bf16_kernel<<<dim3(D_INNER / 64, H_DIM / 64), block, 0, stream>>>(
            ssm_proj, pbfT, H_DIM, D_INNER);
        mamba_inproj_conv_kernel<true><<<grid, block, smem_bytes, stream>>>(
            xbf, pbfT, conv_states, conv_wts, conv_bias, out);
    } else {
        // fallback: convert+transpose during staging (no scratch requirement)
        mamba_inproj_conv_kernel<false><<<grid, block, smem_bytes, stream>>>(
            x, ssm_proj, conv_states, conv_wts, conv_bias, out);
    }
}
